// WaveletSparsityPrior_27676769256021
// MI455X (gfx1250) — matrix operand order repaired and checked by reading, not compiled, not run on hardware
//
#include <hip/hip_runtime.h>
#include <hip/hip_bf16.h>
#include <stdint.h>

// ---------------------------------------------------------------------------
// WaveletSparsityPrior on MI455X (gfx1250, wave32)
//
// Memory-bound: 128MB input, ~0.2 flop/byte -> two streaming passes over HBM
// (~11us at 23.3 TB/s). Compute path uses v_wmma_f32_16x16x32_f16 for the
// separable Haar (Y = W X W^T per 16x16 tile) and global_load_async_to_lds_b128
// for global->LDS staging (ASYNCcnt path).
// ---------------------------------------------------------------------------

typedef __attribute__((ext_vector_type(16))) _Float16 v16h;
typedef __attribute__((ext_vector_type(8)))  float    v8f;

#define HS ((_Float16)0.70710678f)   // 1/sqrt(2): Haar analysis coefficient

// ---------------------------------------------------------------------------
// One 2-D Haar level on a 16x16 f32 tile held in LDS, executed by ONE wave32:
//   D1 = X * W^T   (A = X from LDS, B = W^T synthesized in registers)
//   Y  = W  * D1   (A = W synthesized, B = D1 staged through LDS scratch)
// Layouts per CDNA5 ISA 7.12.2:
//   A 16x32 f16 : lane<16 -> M=lane, elems[0..7]=K0..7 ; lane>=16 -> K8..15
//   B 32x16 f16 : lane<16 -> N=lane, elems[0..7]=K0..7 ; lane>=16 -> K8..15
//   C/D 16x16 f32: vgpr r, lane l -> row = r + 8*(l>>4), col = l&15
// K=16 effective (upper K zero-padded). Intra-wave LDS RAW is safe: DS ops of
// one wave execute in order (ISA 7.3), no barrier needed.
// Returns this lane's sum of min(|detail|, t) over the 3 detail sub-bands;
// writes the 8x8 cA block to dstCA (if non-null).
// ---------------------------------------------------------------------------
__device__ __forceinline__ float haar16_wmma(const float* src, int sstride,
                                             float* scratch,            // 16x16 per-wave
                                             float* dstCA, int dstride, // 8x8 cA out
                                             v16h b1, v16h a2v, float t, int lane)
{
    const int rc    = lane & 15;   // row (A) / col (B,C,D)
    const int khalf = lane >> 4;

    v16h a1 = {};
    #pragma unroll
    for (int i = 0; i < 8; ++i)
        a1[i] = (_Float16)src[rc * sstride + 8 * khalf + i];

    v8f z = {};
    v8f d1 = __builtin_amdgcn_wmma_f32_16x16x32_f16(false, a1, false, b1,
                                                    (short)0, z, false, false);
    #pragma unroll
    for (int r = 0; r < 8; ++r)
        scratch[(r + 8 * khalf) * 16 + rc] = d1[r];

    v16h b2 = {};
    {
        #pragma unroll
        for (int i = 0; i < 8; ++i)
            b2[i] = (_Float16)scratch[(8 * khalf + i) * 16 + rc];
    }
    v8f d2 = __builtin_amdgcn_wmma_f32_16x16x32_f16(false, a2v, false, b2,
                                                    (short)0, z, false, false);
    float s = 0.f;
    #pragma unroll
    for (int r = 0; r < 8; ++r) {
        const int gr = r + 8 * khalf;
        const float y = d2[r];
        if ((gr < 8) && (rc < 8)) {          // cA block
            if (dstCA) dstCA[gr * dstride + rc] = y;
        } else {                             // cH/cV/cD: |c - soft(c,t)| == min(|c|,t)
            s += fminf(fabsf(y), t);
        }
    }
    return s;
}

// ---------------------------------------------------------------------------
// Kernel 1: per-image level-3 |details| -> LDS, exact median (avg of order
// stats 6143/6144 of 12288) via bitwise binary search on float bit patterns,
// then sigma / thr to workspace. One 1024-thread WG per image.
// ---------------------------------------------------------------------------
__global__ __launch_bounds__(1024) void k1_median(const float* __restrict__ pred,
                                                  float* __restrict__ wsSigma,
                                                  float* __restrict__ wsThr,
                                                  float* __restrict__ wsLoss)
{
    __shared__ float absd[12288];   // 48 KB of 320 KB WGP LDS
    __shared__ int   s_cnt;
    const int img = blockIdx.x;
    const int tid = threadIdx.x;
    const float* im = pred + (size_t)img * 512 * 512;

    // Each thread reduces 4 disjoint 8x8 blocks through the cA chain.
    #pragma unroll
    for (int k = 0; k < 4; ++k) {
        const int b  = tid + k * 1024;
        const int bx = b & 63, by = b >> 6;
        const float* bp = im + (by * 8) * 512 + bx * 8;
        float a1[4][4];
        #pragma unroll
        for (int iy = 0; iy < 4; ++iy) {
            const float4 r0a = *(const float4*)(bp + (2 * iy) * 512);
            const float4 r0b = *(const float4*)(bp + (2 * iy) * 512 + 4);
            const float4 r1a = *(const float4*)(bp + (2 * iy + 1) * 512);
            const float4 r1b = *(const float4*)(bp + (2 * iy + 1) * 512 + 4);
            a1[iy][0] = (r0a.x + r0a.y + r1a.x + r1a.y) * 0.5f;
            a1[iy][1] = (r0a.z + r0a.w + r1a.z + r1a.w) * 0.5f;
            a1[iy][2] = (r0b.x + r0b.y + r1b.x + r1b.y) * 0.5f;
            a1[iy][3] = (r0b.z + r0b.w + r1b.z + r1b.w) * 0.5f;
        }
        const float c00 = (a1[0][0] + a1[0][1] + a1[1][0] + a1[1][1]) * 0.5f;
        const float c01 = (a1[0][2] + a1[0][3] + a1[1][2] + a1[1][3]) * 0.5f;
        const float c10 = (a1[2][0] + a1[2][1] + a1[3][0] + a1[3][1]) * 0.5f;
        const float c11 = (a1[2][2] + a1[2][3] + a1[3][2] + a1[3][3]) * 0.5f;
        absd[3 * b + 0] = fabsf((c00 + c01 - c10 - c11) * 0.5f);  // cH3
        absd[3 * b + 1] = fabsf((c00 - c01 + c10 - c11) * 0.5f);  // cV3
        absd[3 * b + 2] = fabsf((c00 - c01 - c10 + c11) * 0.5f);  // cD3
    }
    __syncthreads();

    // Exact k-th order statistic (values >= 0 -> float bits are monotonic).
    uint32_t res[2];
    for (int which = 0; which < 2; ++which) {
        const int k = 6143 + which;
        uint32_t lo = 0u, hi = 0x7F800000u;
        while (lo < hi) {
            const uint32_t mid = lo + ((hi - lo) >> 1);
            if (tid == 0) s_cnt = 0;
            __syncthreads();
            int c = 0;
            for (int i = tid; i < 12288; i += 1024)
                c += (__float_as_uint(absd[i]) <= mid) ? 1 : 0;
            if (c) atomicAdd(&s_cnt, c);
            __syncthreads();
            const int tot = s_cnt;   // uniform
            __syncthreads();
            if (tot >= k + 1) hi = mid; else lo = mid + 1;
        }
        res[which] = lo;
    }
    if (tid == 0) {
        const float med   = 0.5f * (__uint_as_float(res[0]) + __uint_as_float(res[1]));
        const float sigma = med * (1.0f / 0.6745f);
        const float thr   = fminf(fmaxf(sigma * 2.5f, 0.05f), 0.2f); // clip to [BASE/2, 2*BASE]
        wsSigma[img] = sigma;
        wsThr[img]   = thr;
        wsLoss[img]  = 0.f;   // re-zeroed every launch: replay-safe
    }
}

// ---------------------------------------------------------------------------
// Kernel 2: full 3-level DWT + weighted min(|c|,t) sums. One 128-thread WG
// (4 waves) per 64x64 region; region staged via async global->LDS.
// ---------------------------------------------------------------------------
__global__ __launch_bounds__(128) void k2_wmma(const float* __restrict__ pred,
                                               const float* __restrict__ wsThr,
                                               float* __restrict__ wsLoss)
{
    __shared__ float region[64 * 64];   // 16 KB input tile
    __shared__ float ca1[32 * 32];      // level-1 approximation
    __shared__ float ca2[16 * 16];      // level-2 approximation
    __shared__ float scr[4][16 * 16];   // per-wave WMMA staging
    __shared__ float red[128];

    const int tid  = threadIdx.x;
    const int lane = tid & 31;
    const int wv   = tid >> 5;
    const int img  = blockIdx.x >> 6;
    const int reg  = blockIdx.x & 63;
    const int ry   = reg >> 3, rx = reg & 7;
    const float* base = pred + (size_t)img * 512 * 512 + (ry * 64) * 512 + rx * 64;

    // --- async copy: each wave pulls its own 16 rows (2 rows / b128 issue) ---
    {
        const int sub = lane >> 4, c16 = lane & 15;
        #pragma unroll
        for (int i = 0; i < 8; ++i) {
            const int row = wv * 16 + i * 2 + sub;
            const uint32_t lds = (uint32_t)(uintptr_t)&region[row * 64 + c16 * 4];
            const unsigned long long ga =
                (unsigned long long)(uintptr_t)(base + row * 512 + c16 * 4);
            asm volatile("global_load_async_to_lds_b128 %0, %1, off"
                         :: "v"(lds), "v"(ga) : "memory");
        }
        asm volatile("s_wait_asynccnt 0" ::: "memory");
    }
    const float thr = wsThr[img];

    // Per-lane constant operands: b1 = W^T slice (B layout), a2v = W slice (A layout)
    v16h b1 = {};
    {
        const int col = lane & 15;
        const int p = 2 * (col & 7);
        if ((lane >> 4) == (p >> 3)) {
            b1[p & 7]       = HS;
            b1[(p & 7) + 1] = (col < 8) ? HS : (_Float16)(-HS);
        }
    }
    v16h a2v = {};
    {
        const int row = lane & 15, kh = lane >> 4;
        const int i0  = 2 * (row & 7) - 8 * kh;
        if (i0 >= 0 && i0 < 8) {
            a2v[i0]     = HS;
            a2v[i0 + 1] = (row < 8) ? HS : (_Float16)(-HS);
        }
    }

    float acc = 0.f;
    // Level 1 (256x256 details, level_idx=3): t=thr/4, weight 1/(3*3*65536)
    {
        const float t = thr * 0.25f, W1 = 1.f / (3.f * 3.f * 65536.f);
        #pragma unroll
        for (int i = 0; i < 4; ++i) {
            const float* src = region + (wv * 16) * 64 + i * 16;  // wave reads rows it loaded
            float* dst = ca1 + (wv * 8) * 32 + i * 8;
            acc += W1 * haar16_wmma(src, 64, scr[wv], dst, 32, b1, a2v, t, lane);
        }
    }
    __syncthreads();
    // Level 2 (128x128 details, level_idx=2): t=thr/2, weight 1/(3*2*16384)
    {
        const float t = thr * 0.5f, W2 = 1.f / (3.f * 2.f * 16384.f);
        const int tx = wv & 1, ty = wv >> 1;
        const float* src = ca1 + (ty * 16) * 32 + tx * 16;
        float* dst = ca2 + (ty * 8) * 16 + tx * 8;
        acc += W2 * haar16_wmma(src, 32, scr[wv], dst, 16, b1, a2v, t, lane);
    }
    __syncthreads();
    // Level 3 (64x64 details, level_idx=1): t=thr, weight 1/(3*1*4096)
    if (wv == 0) {
        const float t = thr, W3 = 1.f / (3.f * 1.f * 4096.f);
        acc += W3 * haar16_wmma(ca2, 16, scr[0], nullptr, 16, b1, a2v, t, lane);
    }

    // Block reduce, one f32 atomic per block into the per-image accumulator.
    red[tid] = acc;
    __syncthreads();
    for (int off = 64; off > 0; off >>= 1) {
        if (tid < off) red[tid] += red[tid + off];
        __syncthreads();
    }
    if (tid == 0) atomicAdd(&wsLoss[img], red[0]);
}

// ---------------------------------------------------------------------------
// Kernel 3: final reduction over images -> (loss, avg_noise)
// ---------------------------------------------------------------------------
__global__ __launch_bounds__(128) void k3_final(const float* __restrict__ wsLoss,
                                                const float* __restrict__ wsSigma,
                                                float* __restrict__ out, int B)
{
    __shared__ float sl[128], ss[128];
    const int tid = threadIdx.x;
    sl[tid] = (tid < B) ? wsLoss[tid]  : 0.f;
    ss[tid] = (tid < B) ? wsSigma[tid] : 0.f;
    __syncthreads();
    for (int off = 64; off > 0; off >>= 1) {
        if (tid < off) { sl[tid] += sl[tid + off]; ss[tid] += ss[tid + off]; }
        __syncthreads();
    }
    if (tid == 0) {
        out[0] = sl[0] / (float)B;   // loss
        out[1] = ss[0] / (float)B;   // avg_noise
    }
}

extern "C" void kernel_launch(void* const* d_in, const int* in_sizes, int n_in,
                              void* d_out, int out_size, void* d_ws, size_t ws_size,
                              hipStream_t stream)
{
    const float* pred = (const float*)d_in[0];
    const int B = in_sizes[0] / (512 * 512);   // 128 for the reference shapes
    float* ws      = (float*)d_ws;
    float* wsLoss  = ws;         // [128]
    float* wsSigma = ws + 128;   // [128]
    float* wsThr   = ws + 256;   // [128]

    k1_median<<<dim3(B), dim3(1024), 0, stream>>>(pred, wsSigma, wsThr, wsLoss);
    k2_wmma  <<<dim3(B * 64), dim3(128), 0, stream>>>(pred, wsThr, wsLoss);
    k3_final <<<dim3(1), dim3(128), 0, stream>>>(wsLoss, wsSigma, (float*)d_out, B);
}
